// Attention_SIGFormer_30004641530195
// MI455X (gfx1250) — matrix operand order repaired and checked by reading, not compiled, not run on hardware
//
#include <hip/hip_runtime.h>

// MI455X / gfx1250 SIGFormer attention — fused single-pass flash design, v4.
// All hot data L2-resident (xn f32 51MB + combined f16 features 32MB << 192MB L2).
// CSR counting-sort removes scatter atomics. Per-node features packed as one
// contiguous 320-element f16 vector [x*0.25 | SSE]: the 10 WMMA K-chunks load
// from a single base pointer (imm offsets 0..608). A sched_barrier between the
// A-fragment loads and the v_wmma_f32_16x16x32_f16 chain forces all 20 b128
// loads to issue before the first WMMA (one L2 round trip per tile), and
// __launch_bounds__(256,2) raises the VGPR budget so the staging fragments
// stay live. B operand is the row's own vector replicated across all 16
// columns, so each C column equals the 16-edge score vector; dual online
// softmax + aggregation fused with flash-style accumulator rescaling; per-edge
// broadcasts via v_readlane (uniform lane index); no LDS in the hot kernel.

typedef __attribute__((ext_vector_type(16))) _Float16 v16h;
typedef __attribute__((ext_vector_type(8)))  _Float16 v8h;
typedef __attribute__((ext_vector_type(8)))  float    v8f;
typedef __attribute__((ext_vector_type(4)))  float    v4f;

#define DD   256
#define DSS  64
#define FD   320   // combined f16 feature length per node: 256 x + 64 SSE
#define NKC  10    // 8 x-chunks + 2 SSE-chunks of K=32

__device__ __forceinline__ float wave_max32(float v) {
#pragma unroll
  for (int o = 16; o > 0; o >>= 1) v = fmaxf(v, __shfl_xor(v, o, 32));
  return v;
}
__device__ __forceinline__ float wave_sum32(float v) {
#pragma unroll
  for (int o = 16; o > 0; o >>= 1) v += __shfl_xor(v, o, 32);
  return v;
}
__device__ __forceinline__ float readlane_f(float v, int lane) {
  return __int_as_float(__builtin_amdgcn_readlane(__float_as_int(v), lane));
}

// ---- Kernel 1: LayerNorm -> xn (f32) and comb[0..255] (f16, pre-scaled 0.25) ----
__global__ __launch_bounds__(256) void ln_kernel(const float* __restrict__ embs,
                                                 float* __restrict__ xn,
                                                 _Float16* __restrict__ comb, int n) {
  int wv = threadIdx.x >> 5, lane = threadIdx.x & 31;
  int r = blockIdx.x * (blockDim.x >> 5) + wv;
  if (r >= n) return;
  const float* xp = embs + (size_t)r * DD + lane * 8;
  v4f a = *(const v4f*)xp;
  v4f b = *(const v4f*)(xp + 4);
  float vals[8] = {a[0], a[1], a[2], a[3], b[0], b[1], b[2], b[3]};
  float s = 0.f, sq = 0.f;
#pragma unroll
  for (int i = 0; i < 8; ++i) { s += vals[i]; sq += vals[i] * vals[i]; }
  s = wave_sum32(s); sq = wave_sum32(sq);
  float mean = s * (1.0f / DD);
  float var = sq * (1.0f / DD) - mean * mean;
  float rs = rsqrtf(var + 1e-5f);
  float y[8];
  v4f o0, o1;
  v8h hv;
#pragma unroll
  for (int i = 0; i < 8; ++i) {
    y[i] = (vals[i] - mean) * rs;
    hv[i] = (_Float16)(y[i] * 0.25f);   // fold 1/sqrt(D)=1/16 into A*B product
  }
  o0[0] = y[0]; o0[1] = y[1]; o0[2] = y[2]; o0[3] = y[3];
  o1[0] = y[4]; o1[1] = y[5]; o1[2] = y[6]; o1[3] = y[7];
  float* op = xn + (size_t)r * DD + lane * 8;
  *(v4f*)op = o0; *(v4f*)(op + 4) = o1;
  *(v8h*)(comb + (size_t)r * FD + lane * 8) = hv;
}

// ---- Kernel 2: SSE f32 -> comb[256..319] (f16) ----
__global__ void sse_kernel(const float* __restrict__ s, _Float16* __restrict__ comb,
                           int total) {
  int i = blockIdx.x * blockDim.x + threadIdx.x;
  if (i < total) {
    int node = i >> 6, k = i & 63;
    comb[(size_t)node * FD + DD + k] = (_Float16)s[i];
  }
}

// ---- Kernel 3: degree histogram ----
__global__ void count_kernel(const int* __restrict__ spe, int* __restrict__ cnt, int e) {
  int i = blockIdx.x * blockDim.x + threadIdx.x;
  if (i < e) atomicAdd(&cnt[spe[3 * i]], 1);
}

// ---- Kernel 4: single-block exclusive scan -> offs / cursor ----
__global__ void scan_kernel(const int* __restrict__ cnt, int* __restrict__ offs,
                            int* __restrict__ cur, int n) {
  __shared__ int buf[1024];
  int tid = threadIdx.x;
  if (tid == 0) offs[0] = 0;
  int carry = 0;
  for (int base = 0; base < n; base += 1024) {
    int i = base + tid;
    int v = (i < n) ? cnt[i] : 0;
    buf[tid] = v;
    __syncthreads();
    for (int o = 1; o < 1024; o <<= 1) {
      int t = (tid >= o) ? buf[tid - o] : 0;
      __syncthreads();
      buf[tid] += t;
      __syncthreads();
    }
    int incl = buf[tid];
    if (i < n) { offs[i + 1] = carry + incl; cur[i] = carry + incl - v; }
    int tot = buf[1023];
    __syncthreads();
    carry += tot;
  }
}

// ---- Kernel 5: scatter edges into CSR order ----
__global__ void scatter_kernel(const int* __restrict__ spe, int* __restrict__ cur,
                               int* __restrict__ ccol, int* __restrict__ cpt, int e) {
  int i = blockIdx.x * blockDim.x + threadIdx.x;
  if (i < e) {
    int r = spe[3 * i];
    int slot = atomicAdd(&cur[r], 1);
    ccol[slot] = spe[3 * i + 1];
    cpt[slot]  = spe[3 * i + 2];
  }
}

// ---- Kernel 6: fused scores (WMMA) + dual online softmax + aggregation ----
__global__ __launch_bounds__(256, 2) void row_kernel(
    const float* __restrict__ xn, const _Float16* __restrict__ comb,
    const float* __restrict__ pemb, const float* __restrict__ lam,
    const int* __restrict__ offs, const int* __restrict__ ccol,
    const int* __restrict__ cpt, float* __restrict__ out, int n) {
  int wv = threadIdx.x >> 5, lane = threadIdx.x & 31;
  int r = blockIdx.x * (blockDim.x >> 5) + wv;
  if (r >= n) return;
  int start = offs[r];
  int deg = offs[r + 1] - start;
  int h = lane >> 4;        // K-half / M-half selector
  int m = lane & 15;        // edge slot within tile

  float* op = out + (size_t)r * DD + lane * 8;
  if (deg == 0) { v4f z = {}; *(v4f*)op = z; *(v4f*)(op + 4) = z; return; }

  // B operands: this row's combined vector, replicated across all 16 N-columns.
  // B fragment, this code's contiguous k order (same as A): lanes 0-15 K=kc*32+0..15, lanes 16-31 K=+16..31.
  const _Float16* combr = comb + (size_t)r * FD;
  v16h B[NKC];
#pragma unroll
  for (int kc = 0; kc < NKC; ++kc)
    B[kc] = *(const v16h*)(combr + kc * 32 + 16 * h);
  _Float16 lh = (_Float16)lam[0];
  B[8] = B[8] * lh;        // lambda applies only to the SSE chunks
  B[9] = B[9] * lh;

  float m_a = -INFINITY, s_a = 0.f, m_p = -INFINITY, s_p = 0.f;
  v4f accA0 = {}, accA1 = {}, accP0 = {}, accP1 = {};
  int nt = (deg + 15) >> 4;

  for (int t = 0; t < nt; ++t) {
    int base_e = t << 4;
    int slot = base_e + m;
    bool val = slot < deg;
    int cc = val ? ccol[start + slot] : r;   // pad with self (safe address)
    int pt = val ? cpt[start + slot] : 0;

    // ---- issue ALL 20 A-fragment loads before any WMMA consumes one ----
    const _Float16* ap = comb + (size_t)cc * FD;
    v8h alo[NKC], ahi[NKC];
#pragma unroll
    for (int kc = 0; kc < NKC; ++kc) {
      // A fragment, contiguous k order (same as B): lane(m,h): VGPR0-3 = K kc*32+16h+0..7, VGPR4-7 = +8
      alo[kc] = *(const v8h*)(ap + kc * 32 + 16 * h);
      ahi[kc] = *(const v8h*)(ap + kc * 32 + 16 * h + 8);
    }
    __builtin_amdgcn_sched_barrier(0);   // keep loads ahead of the WMMA chain

    // ---- 10-deep WMMA accumulation: C[v] = score(edge v + 8*h), col-replicated ----
    v8f C = {};
#pragma unroll
    for (int kc = 0; kc < NKC; ++kc) {
      v16h A = __builtin_shufflevector(alo[kc], ahi[kc], 0, 1, 2, 3, 4, 5, 6, 7,
                                       8, 9, 10, 11, 12, 13, 14, 15);
      C = __builtin_amdgcn_wmma_f32_16x16x32_f16(false, A, false, B[kc],
                                                 (short)0, C, false, false);
    }

    // ---- tile stats in pure VALU: each lane holds 8 scores (edges v + 8*h) ----
    float lmax = -INFINITY;
#pragma unroll
    for (int v = 0; v < 8; ++v) {
      bool ok = (base_e + v + 8 * h) < deg;
      lmax = fmaxf(lmax, ok ? C[v] : -INFINITY);
    }
    float mna = fmaxf(m_a, wave_max32(lmax));
    float lsum = 0.f;
#pragma unroll
    for (int v = 0; v < 8; ++v) {
      bool ok = (base_e + v + 8 * h) < deg;
      lsum += ok ? __expf(C[v] - mna) : 0.f;
    }
    float ta = wave_sum32((m == 0) ? lsum : 0.f);  // lanes 0/16 carry the halves

    float my_p = pemb[pt];
    bool act = val && (lane < 16);
    float mnp = fmaxf(m_p, wave_max32(act ? my_p : -INFINITY));
    float tp = wave_sum32(act ? __expf(my_p - mnp) : 0.f);

    float scA = (s_a == 0.f) ? 0.f : __expf(m_a - mna);
    float scP = (s_p == 0.f) ? 0.f : __expf(m_p - mnp);
    s_a = s_a * scA + ta;
    s_p = s_p * scP + tp;
    m_a = mna; m_p = mnp;
    accA0 *= scA; accA1 *= scA;
    accP0 *= scP; accP1 *= scP;

    // ---- aggregation: broadcast per-edge scalars via v_readlane (uniform j) ----
    int lim = deg - base_e; if (lim > 16) lim = 16;
#pragma unroll
    for (int j = 0; j < 16; ++j) {
      if (j < lim) {
        float aj = readlane_f(C[j & 7], (j < 8) ? 0 : 16);
        float ea = __expf(aj - mna);
        float pj = readlane_f(my_p, j);
        float ep = __expf(pj - mnp);
        int cj = __builtin_amdgcn_readlane(cc, j);
        const float* gp = xn + (size_t)cj * DD + lane * 8;
        v4f u0 = *(const v4f*)gp;
        v4f u1 = *(const v4f*)(gp + 4);
        accA0 += ea * u0; accA1 += ea * u1;
        accP0 += ep * u0; accP1 += ep * u1;
      }
    }
  }

  float ia = 1.f / s_a, ip = 1.f / s_p;
  v4f o0 = accA0 * ia + accP0 * ip;
  v4f o1 = accA1 * ia + accP1 * ip;
  *(v4f*)op = o0; *(v4f*)(op + 4) = o1;
}

extern "C" void kernel_launch(void* const* d_in, const int* in_sizes, int n_in,
                              void* d_out, int out_size, void* d_ws, size_t ws_size,
                              hipStream_t stream) {
  const float* embs = (const float*)d_in[0];
  const float* sse  = (const float*)d_in[1];
  const int*   spe  = (const int*)d_in[2];
  const float* pemb = (const float*)d_in[3];
  const float* lam  = (const float*)d_in[4];
  int N = in_sizes[0] / DD;
  int E = in_sizes[2] / 3;

  char* ws = (char*)d_ws;
  size_t off = 0;
  auto take = [&](size_t bytes) {
    char* p = ws + off;
    off = (off + bytes + 255) & ~(size_t)255;
    return p;
  };
  float*    xn   = (float*)take((size_t)N * DD * 4);
  _Float16* comb = (_Float16*)take((size_t)N * FD * 2);
  int*      cnt  = (int*)take((size_t)N * 4);
  int*      offs = (int*)take((size_t)(N + 1) * 4);
  int*      cur  = (int*)take((size_t)N * 4);
  int*      ccol = (int*)take((size_t)E * 4);
  int*      cpt  = (int*)take((size_t)E * 4);
  (void)ws_size; (void)n_in; (void)out_size;

  hipMemsetAsync(cnt, 0, (size_t)N * 4, stream);
  ln_kernel<<<(N + 7) / 8, 256, 0, stream>>>(embs, xn, comb, N);
  sse_kernel<<<((N * DSS) + 255) / 256, 256, 0, stream>>>(sse, comb, N * DSS);
  count_kernel<<<(E + 255) / 256, 256, 0, stream>>>(spe, cnt, E);
  scan_kernel<<<1, 1024, 0, stream>>>(cnt, offs, cur, N);
  scatter_kernel<<<(E + 255) / 256, 256, 0, stream>>>(spe, cur, ccol, cpt, E);
  row_kernel<<<(N + 7) / 8, 256, 0, stream>>>(xn, comb, pemb, lam, offs,
                                              ccol, cpt, (float*)d_out, N);
}
